// Linear8bit_89249420411033
// MI455X (gfx1250) — 8-bit matrix operand order repaired and checked by reading, compiled, not run on hardware
//
#include <hip/hip_runtime.h>
#include <stdint.h>


#define M_DIM 8192
#define K_DIM 4096
#define N_DIM 4096
#define THRESH 6.0f

typedef __attribute__((ext_vector_type(8))) int v8i;
typedef __attribute__((ext_vector_type(4))) int v4i;

#if defined(__gfx1250__) && __has_builtin(__builtin_amdgcn_global_load_async_to_lds_b128) && \
    __has_builtin(__builtin_amdgcn_s_wait_asynccnt)
#define USE_ASYNC_LDS 1
#else
#define USE_ASYNC_LDS 0
#endif

#if USE_ASYNC_LDS
#define ASYNC_WAIT(n) __builtin_amdgcn_s_wait_asynccnt(n)
#define GPTR(p) ((__attribute__((address_space(1))) v4i*)(p))
#define LPTR(p) ((__attribute__((address_space(3))) v4i*)(p))
#else
#define ASYNC_WAIT(n)
#endif

static constexpr size_t WS_COLMAX = 0;
static constexpr size_t WS_COUNT  = 16384;
static constexpr size_t WS_LIST   = 16512;
static constexpr size_t WS_SCALEA = 33024;
static constexpr size_t WS_CA     = 65792;

__global__ void init_ws_kernel(unsigned* __restrict__ colmax, int* __restrict__ counter) {
    int i = blockIdx.x * 256 + threadIdx.x;
    if (i < K_DIM) colmax[i] = 0u;
    if (i == 0) *counter = 0;
}

__global__ void col_absmax_kernel(const float* __restrict__ x, unsigned* __restrict__ colmax) {
    const int col  = blockIdx.x * 256 + threadIdx.x;
    const int row0 = blockIdx.y * (M_DIM / 8);
    float mx = 0.0f;
    for (int r = 0; r < M_DIM / 8; ++r)
        mx = fmaxf(mx, fabsf(x[(size_t)(row0 + r) * K_DIM + col]));
    atomicMax(colmax + col, __float_as_uint(mx));
}

__global__ void build_list_kernel(const unsigned* __restrict__ colmax,
                                  int* __restrict__ counter, int* __restrict__ list) {
    if (threadIdx.x == 0 && blockIdx.x == 0) {
        int cnt = 0;
        for (int j = 0; j < K_DIM; ++j)
            if (__uint_as_float(colmax[j]) > THRESH) list[cnt++] = j;
        *counter = cnt;
    }
}

__global__ void row_quant_kernel(const float* __restrict__ x,
                                 const unsigned* __restrict__ colmax,
                                 float* __restrict__ scaleA,
                                 int8_t* __restrict__ cA) {
    const int m   = blockIdx.x;
    const int tid = threadIdx.x;
    float vals[16];
    float amax = 0.0f;
#pragma unroll
    for (int i = 0; i < 16; ++i) {
        const int col = tid + i * 256;
        float v = x[(size_t)m * K_DIM + col];
        const float cm = __uint_as_float(colmax[col]);
        v = (cm > THRESH) ? 0.0f : v;
        vals[i] = v;
        amax = fmaxf(amax, fabsf(v));
    }
    __shared__ float red[256];
    red[tid] = amax;
    __syncthreads();
    for (int s = 128; s > 0; s >>= 1) {
        if (tid < s) red[tid] = fmaxf(red[tid], red[tid + s]);
        __syncthreads();
    }
    const float scale = fmaxf(red[0], 1e-12f);
    if (tid == 0) scaleA[m] = scale;
    const float k = 127.0f / scale;
#pragma unroll
    for (int i = 0; i < 16; ++i) {
        const int col = tid + i * 256;
        cA[(size_t)m * K_DIM + col] = (int8_t)(int)(vals[i] * k);
    }
}

__device__ __forceinline__ v8i load_a_frag(const int8_t* lds, int row, int khalf) {
    const int2* p = (const int2*)(lds + row * 64 + khalf);
    int2 q0 = p[0];
    int2 q1 = p[1];
    int2 q2 = p[4];
    int2 q3 = p[5];
    v8i a;
    a[0] = q0.x; a[1] = q0.y; a[2] = q1.x; a[3] = q1.y;
    a[4] = q2.x; a[5] = q2.y; a[6] = q3.x; a[7] = q3.y;
    return a;
}

__device__ __forceinline__ v8i load_b_frag(const int8_t* lds, int col, int ksel) {
    const int4* p = (const int4*)(lds + col * 64 + ksel * 16);
    int4 b0 = p[0];
    int4 b1 = p[2];
    v8i b;
    b[0] = b0.x; b[1] = b0.y; b[2] = b0.z; b[3] = b0.w;
    b[4] = b1.x; b[5] = b1.y; b[6] = b1.z; b[7] = b1.w;
    return b;
}

__global__ void __launch_bounds__(256)
gemm_int8_wmma_kernel(const int8_t* __restrict__ cA,
                      const int8_t* __restrict__ cw,
                      const float* __restrict__ x,
                      const float* __restrict__ wscale,
                      const float* __restrict__ bias,
                      const float* __restrict__ scaleA,
                      const int* __restrict__ nout_p,
                      const int* __restrict__ lst,
                      float* __restrict__ out) {
    __shared__ __align__(16) int8_t lA[2][128 * 64];
    __shared__ __align__(16) int8_t lB[2][128 * 64];

    const int tid  = threadIdx.x;
    const int lane = tid & 31;
    const int wid  = tid >> 5;
    const int wm   = wid & 3;
    const int wn   = wid >> 2;
    const int mb0  = blockIdx.y * 128;
    const int nb0  = blockIdx.x * 128;

    v8i acc[2][4];
#pragma unroll
    for (int tm = 0; tm < 2; ++tm)
#pragma unroll
        for (int tn = 0; tn < 4; ++tn)
#pragma unroll
            for (int g = 0; g < 8; ++g) acc[tm][tn][g] = 0;

    const int r  = tid >> 2;
    const int ko = (tid & 3) << 4;

    auto issue_copies = [&](int stage, int k0) {
#pragma unroll
        for (int p = 0; p < 2; ++p) {
            const int row = p * 64 + r;
            const int8_t* ga = cA + (size_t)(mb0 + row) * K_DIM + k0 + ko;
            const int8_t* gb = cw + (size_t)(nb0 + row) * K_DIM + k0 + ko;
#if USE_ASYNC_LDS
            __builtin_amdgcn_global_load_async_to_lds_b128(
                GPTR(ga), LPTR(&lA[stage][row * 64 + ko]), 0, 0);
            __builtin_amdgcn_global_load_async_to_lds_b128(
                GPTR(gb), LPTR(&lB[stage][row * 64 + ko]), 0, 0);
#else
            *(int4*)(&lA[stage][row * 64 + ko]) = *(const int4*)ga;
            *(int4*)(&lB[stage][row * 64 + ko]) = *(const int4*)gb;
#endif
        }
    };

    constexpr int NK = K_DIM / 64;
    issue_copies(0, 0);

    for (int kt = 0; kt < NK; ++kt) {
        const int cur = kt & 1;
        if (kt + 1 < NK) {
            issue_copies(cur ^ 1, (kt + 1) * 64);
            ASYNC_WAIT(4);
        } else {
            ASYNC_WAIT(0);
        }
        if (kt + 2 < NK) {
            __builtin_prefetch(cA + (size_t)(mb0 + r) * K_DIM + (kt + 2) * 64 + ko, 0, 0);
            __builtin_prefetch(cw + (size_t)(nb0 + r) * K_DIM + (kt + 2) * 64 + ko, 0, 0);
        }
        __syncthreads();

        v8i af[2], bf[4];
#pragma unroll
        for (int tm = 0; tm < 2; ++tm)
            af[tm] = load_a_frag(&lA[cur][0], wm * 32 + tm * 16 + (lane & 15), (lane >> 4) * 16);
#pragma unroll
        for (int tn = 0; tn < 4; ++tn)
            bf[tn] = load_b_frag(&lB[cur][0], wn * 64 + tn * 16 + (lane & 15), lane >> 4);

#pragma unroll
        for (int tm = 0; tm < 2; ++tm)
#pragma unroll
            for (int tn = 0; tn < 4; ++tn)
                acc[tm][tn] = __builtin_amdgcn_wmma_i32_16x16x64_iu8(
                     true, af[tm],  true, bf[tn],
                    acc[tm][tn],  false,  false);

        __syncthreads();
    }

    const int nOut = *nout_p;
#pragma unroll
    for (int tm = 0; tm < 2; ++tm) {
#pragma unroll
        for (int tn = 0; tn < 4; ++tn) {
            const int n  = nb0 + wn * 64 + tn * 16 + (lane & 15);
            const float sB = wscale[n] * (1.0f / 127.0f);
            const float bv = bias[n];
#pragma unroll
            for (int g = 0; g < 8; ++g) {
                const int m = mb0 + wm * 32 + tm * 16 + g + (lane >> 4) * 8;
                const float sA = scaleA[m] * (1.0f / 127.0f);
                float v = (float)acc[tm][tn][g] * sA * sB;
                float c = 0.0f;
                for (int jj = 0; jj < nOut; ++jj) {
                    const int j = lst[jj];
                    c = fmaf(x[(size_t)m * K_DIM + j],
                             (float)cw[(size_t)n * K_DIM + j] * sB, c);
                }
                out[(size_t)m * N_DIM + n] = v + c + bv;
            }
        }
    }
}

extern "C" void kernel_launch(void* const* d_in, const int* in_sizes, int n_in,
                              void* d_out, int out_size, void* d_ws, size_t ws_size,
                              hipStream_t stream) {
    const float*  x      = (const float*)d_in[0];
    const int8_t* cw     = (const int8_t*)d_in[1];
    const float*  wscale = (const float*)d_in[2];
    const float*  bias   = (const float*)d_in[3];
    float*        out    = (float*)d_out;

    uint8_t* ws = (uint8_t*)d_ws;
    unsigned* colmax  = (unsigned*)(ws + WS_COLMAX);
    int*      counter = (int*)(ws + WS_COUNT);
    int*      list    = (int*)(ws + WS_LIST);
    float*    scaleA  = (float*)(ws + WS_SCALEA);
    int8_t*   cA      = (int8_t*)(ws + WS_CA);

    init_ws_kernel<<<dim3((K_DIM + 256) / 256 + 1), dim3(256), 0, stream>>>(colmax, counter);
    col_absmax_kernel<<<dim3(K_DIM / 256, 8), dim3(256), 0, stream>>>(x, colmax);
    build_list_kernel<<<dim3(1), dim3(1), 0, stream>>>(colmax, counter, list);
    row_quant_kernel<<<dim3(M_DIM), dim3(256), 0, stream>>>(x, colmax, scaleA, cA);
    gemm_int8_wmma_kernel<<<dim3(N_DIM / 128, M_DIM / 128), dim3(256), 0, stream>>>(
        cA, cw, x, wscale, bias, scaleA, counter, list, out);
}
